// FISSA_63058709840319
// MI455X (gfx1250) — matrix operand order repaired and checked by reading, not compiled, not run on hardware
//
#include <hip/hip_runtime.h>
#include <hip/hip_bf16.h>
#include <math.h>

// ---------------------------------------------------------------------------
// Problem constants (match reference)
// ---------------------------------------------------------------------------
#define L_DIM 2048
#define D_DIM 256
#define I_DIM 20000
#define B_BLOCKS 5

typedef __attribute__((ext_vector_type(16))) __bf16    v16bf;
typedef __attribute__((ext_vector_type(8)))  __bf16    v8bf;
typedef __attribute__((ext_vector_type(8)))  float     v8f;
typedef __attribute__((ext_vector_type(4)))  float     f32x4;
typedef __attribute__((ext_vector_type(2)))  unsigned  u32x2;
typedef __attribute__((ext_vector_type(4)))  unsigned  u32x4;
typedef __attribute__((ext_vector_type(4)))  int       i32x4;
typedef __attribute__((ext_vector_type(8)))  int       i32x8;

// TDM availability: device pass only (amdgcn builtins are absent in the host pass).
#if defined(__HIP_DEVICE_COMPILE__) && defined(__has_builtin)
#  if __has_builtin(__builtin_amdgcn_tensor_load_to_lds)
#    define USE_TDM 1
#  endif
#endif
#ifndef USE_TDM
#  define USE_TDM 0
#endif

static __device__ inline unsigned short bfbits(float f) {
    unsigned u = __builtin_bit_cast(unsigned, f);
    u = u + 0x7FFFu + ((u >> 16) & 1u);            // round-to-nearest-even
    return (unsigned short)(u >> 16);
}
static __device__ inline __bf16 f2bf(float f) {
    unsigned short h = bfbits(f);
    return __builtin_bit_cast(__bf16, h);
}
static __device__ inline unsigned packbf2(float a, float b) {
    return (unsigned)bfbits(a) | ((unsigned)bfbits(b) << 16);
}

static __device__ inline float wave_sum(float v) {
#pragma unroll
    for (int o = 16; o > 0; o >>= 1) v += __shfl_xor(v, o, 32);
    return v;
}

// ---------------------------------------------------------------------------
// Tiled WMMA GEMM: C[MT,NT] = epilogue(A[MT,KT] @ op(B)), all dims static.
//   TRANSB=0: B is [KT,NT] row-major;  TRANSB=1: B is [NT,KT] row-major (B^T)
//   EPI: 0=none, 1=*scale, 2=+bias, 3=relu(+bias)
// Block = 128 threads (4 waves). Block tile 64x64, K-step 32.
// Each wave owns a 32x32 region = 2x2 v_wmma_f32_16x16x32_bf16 fragments.
// All dims are exact tile multiples -> no bounds checks, EXEC all-1s.
// ---------------------------------------------------------------------------
template <int MT, int NT, int KT, bool TRANSB, int EPI>
__global__ __launch_bounds__(128)
void gemm_wmma_bf16(const float* __restrict__ A, const float* __restrict__ B,
                    const float* __restrict__ bias, float* __restrict__ C,
                    float scale) {
    // 48-element row pad: 96B rows keep v8bf/v16bf LDS loads 16/32B aligned.
    __shared__ __align__(32) __bf16 As[64][48];   // [m][k]
    __shared__ __align__(32) __bf16 Bs[64][48];   // [n][k]  (B stored K-transposed)

    const int tid  = threadIdx.x;
    const int wave = tid >> 5;
    const int lane = tid & 31;
    const int row0 = blockIdx.y * 64;
    const int col0 = blockIdx.x * 64;
    const int wm   = (wave >> 1) * 32;   // wave row offset inside block tile
    const int wn   = (wave & 1) * 32;    // wave col offset inside block tile

    v8f acc[2][2] = {};

    const int kh   = lane >> 4;          // K-half selector (contiguous K order, same for A and B)
    const int mrow = lane & 31 & 15;

    for (int k0 = 0; k0 < KT; k0 += 32) {
        // prefetch next K-tile of the A stream (-> global_prefetch_b8)
        if (k0 + 32 < KT)
            __builtin_prefetch(A + (size_t)(row0 + (tid & 63)) * KT + (k0 + 32), 0, 3);

        __syncthreads();
        // ---- stage A tile 64x32: float4 load -> pack 2x bf16 -> b64 LDS store
#pragma unroll
        for (int j = 0; j < 4; j++) {
            int e  = tid + j * 128;      // 512 float4 chunks
            int r  = e >> 3;             // 8 chunks per 32-wide row
            int c4 = (e & 7) * 4;
            f32x4 v = *(const f32x4*)(A + (size_t)(row0 + r) * KT + (k0 + c4));
            u32x2 p; p.x = packbf2(v.x, v.y); p.y = packbf2(v.z, v.w);
            *(u32x2*)(&As[r][c4]) = p;
        }
        // ---- stage B tile as [n][k] ----
        if (TRANSB) {
#pragma unroll
            for (int j = 0; j < 4; j++) {
                int e  = tid + j * 128;
                int r  = e >> 3;
                int c4 = (e & 7) * 4;
                f32x4 v = *(const f32x4*)(B + (size_t)(col0 + r) * KT + (k0 + c4));
                u32x2 p; p.x = packbf2(v.x, v.y); p.y = packbf2(v.z, v.w);
                *(u32x2*)(&Bs[r][c4]) = p;
            }
        } else {
#pragma unroll
            for (int j = 0; j < 4; j++) {
                int e  = tid + j * 128;  // 512 float4 chunks of the [32 x 64] slab
                int k  = e >> 4;         // 16 chunks per 64-wide row
                int n4 = (e & 15) * 4;
                f32x4 v = *(const f32x4*)(B + (size_t)(k0 + k) * NT + (col0 + n4));
                Bs[n4 + 0][k] = f2bf(v.x);
                Bs[n4 + 1][k] = f2bf(v.y);
                Bs[n4 + 2][k] = f2bf(v.z);
                Bs[n4 + 3][k] = f2bf(v.w);
            }
        }
        __syncthreads();

        // ---- load fragments (contiguous K order: K = kh*16 + 0..15 for A and B) ----
        v16bf afrag[2], bfrag[2];
#pragma unroll
        for (int mi = 0; mi < 2; mi++) {
            const __bf16* ap = &As[wm + mi * 16 + mrow][0];
            afrag[mi] = *(const v16bf*)(ap + kh * 16);    // K = kh*16 + 0..15
        }
#pragma unroll
        for (int ni = 0; ni < 2; ni++) {
            bfrag[ni] = *(const v16bf*)(&Bs[wn + ni * 16 + mrow][kh * 16]); // K = kh*16 + 0..15
        }
#pragma unroll
        for (int mi = 0; mi < 2; mi++)
#pragma unroll
            for (int ni = 0; ni < 2; ni++)
                acc[mi][ni] = __builtin_amdgcn_wmma_f32_16x16x32_bf16(
                    false, afrag[mi], false, bfrag[ni],
                    (short)0, acc[mi][ni], false, false);
    }

    // ---- epilogue + store (C frag: M = i + 8*(lane>>4), N = lane&15) ----
    const int hh = lane >> 4;
#pragma unroll
    for (int mi = 0; mi < 2; mi++)
#pragma unroll
        for (int ni = 0; ni < 2; ni++) {
            const int gcol = col0 + wn + ni * 16 + (lane & 15);
            float* cptr = C + (size_t)(row0 + wm + mi * 16 + 8 * hh) * NT + gcol;
            float bv = (EPI == 2 || EPI == 3) ? bias[gcol] : 0.0f;
#pragma unroll
            for (int i = 0; i < 8; i++) {
                float v = acc[mi][ni][i];
                if (EPI == 1) v *= scale;
                if (EPI == 2 || EPI == 3) v += bv;
                if (EPI == 3) v = fmaxf(v, 0.0f);
                cptr[(size_t)i * NT] = v;   // static stride -> immediate offsets
            }
        }
}

// ---------------------------------------------------------------------------
// Row softmax with CDNA5 async global->LDS staging (ASYNCcnt path).
// One 256-thread block per row of length 2048 (8 KB staged, 1R+1W of HBM).
// ---------------------------------------------------------------------------
__global__ __launch_bounds__(256)
void softmax_rows_async_kernel(float* __restrict__ A, int n) {
    __shared__ __align__(16) float buf[2048];
    __shared__ float red[256];
    float* row = A + (size_t)blockIdx.x * n;
    int t = threadIdx.x;

    // stage 8 floats per thread = 2x 16B async chunks
    {
        unsigned           lds = (unsigned)(unsigned long long)&buf[t * 8];
        unsigned long long g   = (unsigned long long)(row + t * 8);
        asm volatile(
            "global_load_async_to_lds_b128 %0, %1, off\n\t"
            "global_load_async_to_lds_b128 %0, %1, off offset:16\n\t"
            "s_wait_asynccnt 0x0"
            :: "v"(lds), "v"(g) : "memory");
    }
    __syncthreads();

    float mx = -3.0e38f;
    for (int i = t; i < n; i += 256) mx = fmaxf(mx, buf[i]);
    red[t] = mx; __syncthreads();
    for (int s = 128; s > 0; s >>= 1) { if (t < s) red[t] = fmaxf(red[t], red[t + s]); __syncthreads(); }
    mx = red[0]; __syncthreads();

    float sum = 0.f;
    for (int i = t; i < n; i += 256) { float e = __expf(buf[i] - mx); buf[i] = e; sum += e; }
    red[t] = sum; __syncthreads();
    for (int s = 128; s > 0; s >>= 1) { if (t < s) red[t] += red[t + s]; __syncthreads(); }
    float inv = 1.0f / red[0];
    __syncthreads();

    for (int i = t; i < n; i += 256) row[i] = buf[i] * inv;
}

// ---------------------------------------------------------------------------
// Column prefix-sum with Tensor Data Mover double buffering.
// out[r,c] = sum_{r'<=r} in[r',c], in = [2048 x 256] f32.
// Wave 0 issues TDM loads of 64x256 f32 tiles (64 KB) into a 2x64KB LDS ring
// (TENSORcnt), overlapping DMA of chunk c+1 with the scan of chunk c.
// ---------------------------------------------------------------------------
#define SCAN_ROWS 64

#if USE_TDM
static __device__ inline void tdm_issue_2d(const float* gsrc, unsigned lds_addr,
                                           int tensor_rows) {
    unsigned long long ga = (unsigned long long)gsrc;
    u32x4 g0;
    g0.x = 1u;                                             // count=1 valid descriptor
    g0.y = lds_addr;                                       // D#.lds_addr
    g0.z = (unsigned)(ga & 0xFFFFFFFFull);                 // global_addr[31:0]
    g0.w = (unsigned)((ga >> 32) & 0x1FFFFFFull)           // global_addr[56:32]
         | (2u << 30);                                     // type=2 ("image")
    i32x8 g1;
    g1[0] = (2 << 16);                                     // wg_mask=0 | data_size=4B
    g1[1] = (int)((unsigned)(D_DIM & 0xFFFF) << 16);       // tensor_dim0 lo16 (=256)
    g1[2] = (int)(((unsigned)tensor_rows & 0xFFFFu) << 16);// dim0 hi16 | tensor_dim1 lo16
    g1[3] = (int)((unsigned)(D_DIM & 0xFFFF) << 16);       // dim1 hi16 | tile_dim0=256
    g1[4] = SCAN_ROWS;                                     // tile_dim1=64 | tile_dim2=0
    g1[5] = D_DIM;                                         // tensor_dim0_stride lo32
    g1[6] = 0;                                             // stride0 hi16 | stride1 lo16
    g1[7] = 0;                                             // stride1 hi32
    i32x4 z4 = (i32x4)0;
#if __clang_major__ >= 23
    i32x8 z8 = (i32x8)0;
    __builtin_amdgcn_tensor_load_to_lds(g0, g1, z4, z4, z8, 0);
#else
    __builtin_amdgcn_tensor_load_to_lds(g0, g1, z4, z4, 0);
#endif
}
#endif

__global__ __launch_bounds__(256)
void colscan_kernel(const float* __restrict__ in, float* __restrict__ out,
                    int rows, int cols) {
#if USE_TDM
    __shared__ __align__(16) float tbuf[2][SCAN_ROWS * D_DIM];   // 128 KB ring
    const int t = threadIdx.x;
    const int nchunks = rows / SCAN_ROWS;

    if (t < 32)   // one wave issues the DMA (TDM ignores EXEC; issue once per WG)
        tdm_issue_2d(in, (unsigned)(unsigned long long)&tbuf[0][0], rows);

    float run = 0.f;
    for (int c = 0; c < nchunks; ++c) {
        if (t < 32) {
            if (c + 1 < nchunks) {
                tdm_issue_2d(in + (size_t)(c + 1) * SCAN_ROWS * D_DIM,
                             (unsigned)(unsigned long long)&tbuf[(c + 1) & 1][0], rows);
                __builtin_amdgcn_s_wait_tensorcnt(1);   // chunk c landed
            } else {
                __builtin_amdgcn_s_wait_tensorcnt(0);
            }
        }
        __syncthreads();                                 // publish LDS tile
        const float* buf = &tbuf[c & 1][0];
        size_t gbase = (size_t)c * SCAN_ROWS * D_DIM + t;
        for (int r = 0; r < SCAN_ROWS; ++r) {
            run += buf[r * D_DIM + t];
            out[gbase + (size_t)r * D_DIM] = run;
        }
        __syncthreads();                                 // done reading before reuse
    }
#else
    // host-pass / no-TDM fallback: direct streaming scan
    int c = threadIdx.x;
    float run = 0.f;
    for (int r = 0; r < rows; r++) {
        run += in[(size_t)r * cols + c];
        out[(size_t)r * cols + c] = run;
    }
#endif
}

// ---------------------------------------------------------------------------
// Small helper kernels
// ---------------------------------------------------------------------------

// out[row] = dot(q, Rows[row,:])   (one wave per row, 8 rows per block)
__global__ __launch_bounds__(256)
void rowdot_kernel(const float* __restrict__ q, const float* __restrict__ Rows,
                   float* __restrict__ out, int nrows, int d) {
    int wave = threadIdx.x >> 5, lane = threadIdx.x & 31;
    int row  = blockIdx.x * 8 + wave;
    if (row >= nrows) return;
    float s = 0.f;
    for (int c = lane; c < d; c += 32) s += q[c] * Rows[(size_t)row * d + c];
    s = wave_sum(s);
    if (lane == 0) out[row] = s;
}

// out[c] = sum_r w[r] * Mat[r, c]   (one block per column)
__global__ __launch_bounds__(256)
void weighted_colsum_kernel(const float* __restrict__ w, const float* __restrict__ Mat,
                            float* __restrict__ out, int rows, int cols) {
    __shared__ float red[256];
    int c = blockIdx.x, t = threadIdx.x;
    float s = 0.f;
    for (int r = t; r < rows; r += 256) s += w[r] * Mat[(size_t)r * cols + c];
    red[t] = s; __syncthreads();
    for (int st = 128; st > 0; st >>= 1) { if (t < st) red[t] += red[t + st]; __syncthreads(); }
    if (t == 0) out[c] = red[0];
}

__global__ void add_kernel(const float* __restrict__ a, const float* __restrict__ b,
                           float* __restrict__ c, int n) {
    int i = blockIdx.x * 256 + threadIdx.x;
    if (i < n) c[i] = a[i] + b[i];
}

// scal = { dot(m,Wg[0:d]), dot(y,Wg[d:2d]), dot(y,m), bg }   (d==256, 1 block)
__global__ __launch_bounds__(256)
void scalars_kernel(const float* __restrict__ m, const float* __restrict__ y,
                    const float* __restrict__ Wg, const float* __restrict__ bg,
                    float* __restrict__ scal) {
    __shared__ float r1[256], r2[256], r3[256];
    int t = threadIdx.x;
    r1[t] = m[t] * Wg[t];
    r2[t] = y[t] * Wg[D_DIM + t];
    r3[t] = y[t] * m[t];
    __syncthreads();
    for (int s = 128; s > 0; s >>= 1) {
        if (t < s) { r1[t] += r1[t + s]; r2[t] += r2[t + s]; r3[t] += r3[t + s]; }
        __syncthreads();
    }
    if (t == 0) { scal[0] = r1[0]; scal[1] = r2[0]; scal[2] = r3[0]; scal[3] = bg[0]; }
}

// R[l] = g*dot(X[l],m) + (1-g)*dot(y,m);  g = sigmoid(dot(E[l],Wg[2d:3d]) + c1 + c2 + bg)
__global__ __launch_bounds__(256)
void final_gate_kernel(const float* __restrict__ E, const float* __restrict__ X,
                       const float* __restrict__ Wg, const float* __restrict__ m,
                       const float* __restrict__ scal, float* __restrict__ R,
                       int nrows, int d) {
    int wave = threadIdx.x >> 5, lane = threadIdx.x & 31;
    int row  = blockIdx.x * 8 + wave;
    if (row >= nrows) return;
    float s1 = 0.f, s2 = 0.f;
    size_t base = (size_t)row * d;
    for (int c = lane; c < d; c += 32) {
        s1 += E[base + c] * Wg[2 * d + c];
        s2 += X[base + c] * m[c];
    }
    s1 = wave_sum(s1);
    s2 = wave_sum(s2);
    if (lane == 0) {
        float g = 1.0f / (1.0f + __expf(-(s1 + scal[0] + scal[1] + scal[3])));
        R[row] = g * s2 + (1.0f - g) * scal[2];
    }
}

// ---------------------------------------------------------------------------
// Launch
// ---------------------------------------------------------------------------
extern "C" void kernel_launch(void* const* d_in, const int* in_sizes, int n_in,
                              void* d_out, int out_size, void* d_ws, size_t ws_size,
                              hipStream_t stream) {
    const float* S    = (const float*)d_in[0];   // [L, I]
    const float* cand = (const float*)d_in[1];   // [1, I]
    const float* Mw   = (const float*)d_in[2];   // [I, d]
    const float* P    = (const float*)d_in[3];   // [L, d]
    const float* qS   = (const float*)d_in[4];   // [1, d]
    const float* Wg   = (const float*)d_in[5];   // [3d, 1]
    const float* bg   = (const float*)d_in[6];   // [1, 1]
    const float* Q    = (const float*)d_in[7];
    const float* K    = (const float*)d_in[8];
    const float* V    = (const float*)d_in[9];
    const float* Wk   = (const float*)d_in[10];
    const float* Wv   = (const float*)d_in[11];
    const float* W1   = (const float*)d_in[12];
    const float* W2   = (const float*)d_in[13];
    const float* b1   = (const float*)d_in[14];
    const float* b2   = (const float*)d_in[15];
    float* R = (float*)d_out;                    // [L, 1]

    const int L = L_DIM, d = D_DIM, I = I_DIM;
    const size_t LD = (size_t)L * d;

    float* ws = (float*)d_ws;
    float* E      = ws;            // L*d
    float* EWk    = E      + LD;
    float* EWv    = EWk    + LD;
    float* X      = EWv    + LD;
    float* XQ     = X      + LD;
    float* XK     = XQ     + LD;
    float* XV     = XK     + LD;
    float* CV     = XV     + LD;
    float* SAL    = CV     + LD;
    float* H      = SAL    + LD;
    float* Scores = H      + LD;                  // L*L
    float* attn_g = Scores + (size_t)L * L;       // L
    float* yv     = attn_g + L;                   // d
    float* mv     = yv     + d;                   // d
    float* scal   = mv     + d;                   // 4

    dim3 gLD(d / 64, L / 64);   // [L,d] GEMM grid
    dim3 gLL(L / 64, L / 64);   // [L,L] GEMM grid
    const float inv_sqrt_d = 0.0625f;   // 1/sqrt(256)

    // E = S @ M      (dominant, bandwidth-bound: ~184 MB streamed once)
    gemm_wmma_bf16<L_DIM, D_DIM, I_DIM, false, 0><<<gLD, 128, 0, stream>>>(S, Mw, nullptr, E, 1.f);

    // Global item-similarity attention -> y
    gemm_wmma_bf16<L_DIM, D_DIM, D_DIM, false, 0><<<gLD, 128, 0, stream>>>(E, Wk, nullptr, EWk, 1.f);
    gemm_wmma_bf16<L_DIM, D_DIM, D_DIM, false, 0><<<gLD, 128, 0, stream>>>(E, Wv, nullptr, EWv, 1.f);
    rowdot_kernel<<<L / 8, 256, 0, stream>>>(qS, EWk, attn_g, L, d);
    softmax_rows_async_kernel<<<1, 256, 0, stream>>>(attn_g, L);
    weighted_colsum_kernel<<<d, 256, 0, stream>>>(attn_g, EWv, yv, L, d);

    // X = E + P
    add_kernel<<<(int)(LD / 256), 256, 0, stream>>>(E, P, X, (int)LD);

    // Transformer blocks.  (A @ tril) @ XV  ==  A @ cumsum(XV, rows)
    for (int b = 0; b < B_BLOCKS; b++) {
        gemm_wmma_bf16<L_DIM, D_DIM, D_DIM, false, 0><<<gLD, 128, 0, stream>>>(X, Q, nullptr, XQ, 1.f);
        gemm_wmma_bf16<L_DIM, D_DIM, D_DIM, false, 0><<<gLD, 128, 0, stream>>>(X, K, nullptr, XK, 1.f);
        gemm_wmma_bf16<L_DIM, D_DIM, D_DIM, false, 0><<<gLD, 128, 0, stream>>>(X, V, nullptr, XV, 1.f);
        colscan_kernel<<<1, 256, 0, stream>>>(XV, CV, L, d);
        gemm_wmma_bf16<L_DIM, L_DIM, D_DIM, true, 1><<<gLL, 128, 0, stream>>>(XQ, XK, nullptr, Scores, inv_sqrt_d);
        softmax_rows_async_kernel<<<L, 256, 0, stream>>>(Scores, L);
        gemm_wmma_bf16<L_DIM, D_DIM, L_DIM, false, 0><<<gLD, 128, 0, stream>>>(Scores, CV, nullptr, SAL, 1.f);
        gemm_wmma_bf16<L_DIM, D_DIM, D_DIM, false, 3><<<gLD, 128, 0, stream>>>(SAL, W1, b1, H, 1.f);
        gemm_wmma_bf16<L_DIM, D_DIM, D_DIM, false, 2><<<gLD, 128, 0, stream>>>(H, W2, b2, X, 1.f);
    }

    // m = candidate @ M ; scalar constants ; fused gate + readout
    weighted_colsum_kernel<<<d, 256, 0, stream>>>(cand, Mw, mv, I, d);
    scalars_kernel<<<1, 256, 0, stream>>>(mv, yv, Wg, bg, scal);
    final_gate_kernel<<<L / 8, 256, 0, stream>>>(E, X, Wg, mv, scal, R, L, d);
}
